// BilinearlyModulatedAttention_25048249270595
// MI455X (gfx1250) — compile-verified, matrix operand order checked by reading, not run on hardware
//
#include <hip/hip_runtime.h>
#include <hip/hip_bf16.h>
#include <math.h>

typedef __bf16 bf16_t;
typedef __attribute__((ext_vector_type(16))) __bf16 v16bf;
typedef __attribute__((ext_vector_type(8)))  __bf16 v8bf;
typedef __attribute__((ext_vector_type(8)))  float  v8f;

#define D_MODEL 1024
#define N_HEADS 16
#define D_HEAD  64
#define BATCH   2
#define SEQ     2048
#define TOKENS  (BATCH * SEQ)      // 4096
#define QKV_N   (3 * D_MODEL)      // 3072
#define KSTEP   32

// ---------------------------------------------------------------------------
// WMMA helpers
// ---------------------------------------------------------------------------
__device__ __forceinline__ v8f wmma_bf16(v16bf a, v16bf b, v8f c) {
    return __builtin_amdgcn_wmma_f32_16x16x32_bf16(
        false, a, false, b, (short)0, c, false, false);
}

// Load a 16x32 A-fragment (or B^T-fragment) from a row-major bf16 matrix.
//   lanes 0-15 : row = lane,     K = {0..7, 16..23}
//   lanes 16-31: row = lane-16,  K = {8..15, 24..31}
__device__ __forceinline__ v16bf load_frag(const bf16_t* p, int ld) {
    int lane = threadIdx.x & 31;
    const bf16_t* base = p + (lane & 15) * ld + (lane >> 4) * 8;
    v8bf lo = *(const v8bf*)(base);
    v8bf hb = *(const v8bf*)(base + 16);
    return __builtin_shufflevector(lo, hb, 0, 1, 2, 3, 4, 5, 6, 7,
                                           8, 9, 10, 11, 12, 13, 14, 15);
}

__device__ __forceinline__ float rowmax16(float v) {
#pragma unroll
    for (int m = 1; m < 16; m <<= 1) v = fmaxf(v, __shfl_xor(v, m, 32));
    return v;
}
__device__ __forceinline__ float rowsum16(float v) {
#pragma unroll
    for (int m = 1; m < 16; m <<= 1) v += __shfl_xor(v, m, 32);
    return v;
}

// ---------------------------------------------------------------------------
// CDNA5 async copy: global -> LDS, 16B per lane, tracked by ASYNCcnt.
// Block copies a 64x32 bf16 B-tile (4KB) with 256 lanes.
// ---------------------------------------------------------------------------
__device__ __forceinline__ void issue_b_copy(const bf16_t* __restrict__ WT, int ldw,
                                             int col0, int k0, bf16_t* ldsbuf) {
    int t = threadIdx.x;                 // 0..255
    int r = t >> 2;                      // row of WT (output column), 0..63
    int c = (t & 3) * 8;                 // bf16 element offset: 0,8,16,24
    const bf16_t* g = WT + (size_t)(col0 + r) * ldw + k0 + c;
    unsigned lds_off = (unsigned)(size_t)(ldsbuf + r * KSTEP + c);
    asm volatile("global_load_async_to_lds_b128 %0, %1, off"
                 :: "v"(lds_off), "v"(g) : "memory");
}
__device__ __forceinline__ void wait_async_0() {
    asm volatile("s_wait_asynccnt 0x0" ::: "memory");
}

// Pin the pipeline shape: issue all 8 DS reads, then the 4 WMMAs, so the
// DS-counter waits become a partial descending chain instead of 4x dscnt==0.
__device__ __forceinline__ void sched_ds_then_wmma() {
#if __has_builtin(__builtin_amdgcn_sched_group_barrier)
    __builtin_amdgcn_sched_group_barrier(0x100, 8, 0);   // 8 DS reads
    __builtin_amdgcn_sched_group_barrier(0x008, 4, 0);   // 4 MFMA/WMMA
#endif
}

__device__ __forceinline__ void do_step(v8f acc[4], v16bf a, const bf16_t* lds) {
    v16bf b0 = load_frag(lds + 0 * 16 * KSTEP, KSTEP);
    v16bf b1 = load_frag(lds + 1 * 16 * KSTEP, KSTEP);
    v16bf b2 = load_frag(lds + 2 * 16 * KSTEP, KSTEP);
    v16bf b3 = load_frag(lds + 3 * 16 * KSTEP, KSTEP);
    acc[0] = wmma_bf16(a, b0, acc[0]);
    acc[1] = wmma_bf16(a, b1, acc[1]);
    acc[2] = wmma_bf16(a, b2, acc[2]);
    acc[3] = wmma_bf16(a, b3, acc[3]);
    sched_ds_then_wmma();
}

// ---------------------------------------------------------------------------
// Shared GEMM core, manually unrolled x2 over the double buffer so the
// accumulators stay pinned; ONE barrier per K-step:
//   wait_asynccnt 0 -> barrier (publish cur tile + prove other tile drained)
//   -> issue async copy of next tile -> prefetch next A -> 4 WMMAs.
// ---------------------------------------------------------------------------
__device__ __forceinline__ void gemm_core(const bf16_t* __restrict__ A,
                                          const bf16_t* __restrict__ WT,
                                          int K, int row0, int col0,
                                          bf16_t* lds0, bf16_t* lds1, v8f acc[4]) {
    const int nsteps = K / KSTEP;                 // even (K = 1024)
    const bf16_t* Arow = A + (size_t)row0 * K;
    issue_b_copy(WT, K, col0, 0, lds0);
    v16bf a0 = load_frag(Arow, K);
    v16bf a1;
    for (int s = 0; s < nsteps; s += 2) {
        // ---- even step: consume lds0 ----
        wait_async_0();
        __syncthreads();
        if (s + 1 < nsteps) {
            issue_b_copy(WT, K, col0, (s + 1) * KSTEP, lds1);
            a1 = load_frag(Arow + (s + 1) * KSTEP, K);
        }
        do_step(acc, a0, lds0);
        if (s + 1 >= nsteps) break;
        // ---- odd step: consume lds1 ----
        wait_async_0();
        __syncthreads();
        if (s + 2 < nsteps) {
            issue_b_copy(WT, K, col0, (s + 2) * KSTEP, lds0);
            a0 = load_frag(Arow + (s + 2) * KSTEP, K);
        }
        do_step(acc, a1, lds1);
    }
}

// ---------------------------------------------------------------------------
// Conversion / transpose kernels (tiny cost; everything lives in L2)
// ---------------------------------------------------------------------------
__global__ void cvt_f32_bf16(const float* __restrict__ src,
                             bf16_t* __restrict__ dst, int n) {
    int i = blockIdx.x * blockDim.x + threadIdx.x;
    if (i < n) dst[i] = (bf16_t)src[i];
}

__global__ void transpose_bf16(const float* __restrict__ src,
                               bf16_t* __restrict__ dst, int rows, int cols) {
    int i = blockIdx.x * blockDim.x + threadIdx.x;
    if (i < rows * cols) {
        int c = i / rows, r = i - c * rows;
        dst[i] = (bf16_t)src[r * cols + c];
    }
}

__global__ void wg_transpose(const float* __restrict__ src,
                             bf16_t* __restrict__ dst) {
    int i = blockIdx.x * blockDim.x + threadIdx.x;
    if (i < N_HEADS * D_HEAD * D_HEAD) {
        int h = i >> 12, rem = i & 4095;
        int e = rem >> 6, d = rem & 63;
        dst[i] = (bf16_t)src[(h << 12) + d * D_HEAD + e];
    }
}

// ---------------------------------------------------------------------------
// QKV projection: (4096x1024) x (1024x3072) + bias.
// ---------------------------------------------------------------------------
__global__ void __launch_bounds__(256)
qkv_gemm_kernel(const bf16_t* __restrict__ Xb, const bf16_t* __restrict__ WT,
                const float* __restrict__ bias,
                bf16_t* __restrict__ Qb, bf16_t* __restrict__ Kb,
                bf16_t* __restrict__ Vb) {
    __shared__ __align__(16) bf16_t Bt0[64 * KSTEP];
    __shared__ __align__(16) bf16_t Bt1[64 * KSTEP];
    int wv   = threadIdx.x >> 5;
    int lane = threadIdx.x & 31;
    int hi   = lane >> 4;
    int row0 = blockIdx.y * 128 + wv * 16;
    int col0 = blockIdx.x * 64;

    v8f acc[4] = {{}, {}, {}, {}};
    gemm_core(Xb, WT, D_MODEL, row0, col0, Bt0, Bt1, acc);

#pragma unroll
    for (int t = 0; t < 4; ++t) {
        int col    = col0 + t * 16 + (lane & 15);
        int which  = col >> 10;           // 0:Q 1:K 2:V
        int within = col & 1023;
        int head   = within >> 6;
        int feat   = within & 63;
        bf16_t* dst = (which == 0) ? Qb : (which == 1) ? Kb : Vb;
        float bv = bias[col];
#pragma unroll
        for (int i = 0; i < 8; ++i) {
            int token = row0 + i + 8 * hi;
            int bb = token >> 11, tt = token & (SEQ - 1);
            dst[(((bb * N_HEADS + head) * SEQ) + tt) * D_HEAD + feat] =
                (bf16_t)(acc[t][i] + bv);
        }
    }
}

// ---------------------------------------------------------------------------
// Gate: g = sigmoid(q @ W_g[h]);  Vg^T[b,h,feat,t] = g * v
// ---------------------------------------------------------------------------
__global__ void __launch_bounds__(256)
gate_kernel(const bf16_t* __restrict__ Qb, const bf16_t* __restrict__ Vb,
            const bf16_t* __restrict__ WgT, bf16_t* __restrict__ VgT) {
    int wv   = threadIdx.x >> 5;
    int lane = threadIdx.x & 31;
    int hi   = lane >> 4;
    int wg   = blockIdx.x * 8 + wv;
    int bh   = wg >> 7;
    int qt   = wg & 127;
    int h    = bh & (N_HEADS - 1);
    int t0   = qt * 16;

    const bf16_t* Qh = Qb + ((size_t)bh * SEQ + t0) * D_HEAD;
    v16bf qa0 = load_frag(Qh, D_HEAD);
    v16bf qa1 = load_frag(Qh + 32, D_HEAD);

#pragma unroll
    for (int t = 0; t < 4; ++t) {
        v8f acc = {};
        const bf16_t* Wr = WgT + (h << 12) + (t * 16) * D_HEAD;
        acc = wmma_bf16(qa0, load_frag(Wr, D_HEAD), acc);
        acc = wmma_bf16(qa1, load_frag(Wr + 32, D_HEAD), acc);
        int feat = t * 16 + (lane & 15);
        v8bf pack;
#pragma unroll
        for (int i = 0; i < 8; ++i) {
            int tok = t0 + i + 8 * hi;
            float g = 1.0f / (1.0f + __expf(-acc[i]));
            float v = (float)Vb[((size_t)bh * SEQ + tok) * D_HEAD + feat];
            pack[i] = (bf16_t)(g * v);
        }
        *(v8bf*)(VgT + ((size_t)bh * D_HEAD + feat) * SEQ + t0 + 8 * hi) = pack;
    }
}

// ---------------------------------------------------------------------------
// Causal flash attention: one wave per (b,h,16-query tile).
// ---------------------------------------------------------------------------
__global__ void __launch_bounds__(256)
BilinearlyModulatedAttention_flash_kernel(
        const bf16_t* __restrict__ Qb, const bf16_t* __restrict__ Kb,
        const bf16_t* __restrict__ VgT, bf16_t* __restrict__ Oa) {
    __shared__ __align__(16) bf16_t plds[8][16 * 32];
    int wv   = threadIdx.x >> 5;
    int lane = threadIdx.x & 31;
    int hi   = lane >> 4;
    int wg   = blockIdx.x * 8 + wv;
    int bh   = wg >> 7;
    int qt   = wg & 127;
    int b    = bh >> 4;
    int h    = bh & (N_HEADS - 1);
    bf16_t* pw = plds[wv];

    const bf16_t* Qh = Qb  + ((size_t)bh * SEQ + qt * 16) * D_HEAD;
    const bf16_t* Kh = Kb  +  (size_t)bh * SEQ * D_HEAD;
    const bf16_t* Vh = VgT +  (size_t)bh * D_HEAD * SEQ;

    v16bf qa0 = load_frag(Qh, D_HEAD);
    v16bf qa1 = load_frag(Qh + 32, D_HEAD);

    v8f o[4] = {{}, {}, {}, {}};
    float mrun[8], lrun[8];
#pragma unroll
    for (int i = 0; i < 8; ++i) { mrun[i] = -__builtin_inff(); lrun[i] = 0.0f; }

    const float scale = 0.125f;                    // 1/sqrt(64)
    int nkb = (qt + 2) >> 1;                       // ceil((qt+1)*16 / 32)
    int qrowbase = qt * 16 + 8 * hi;
    int n15 = lane & 15;

    for (int kb = 0; kb < nkb; ++kb) {
        int kbase = kb * 32;
        // ---- scores ----
        v8f s0 = {}, s1 = {};
        const bf16_t* K0 = Kh + (size_t)kbase * D_HEAD;
        s0 = wmma_bf16(qa0, load_frag(K0, D_HEAD), s0);
        s0 = wmma_bf16(qa1, load_frag(K0 + 32, D_HEAD), s0);
        const bf16_t* K1 = Kh + (size_t)(kbase + 16) * D_HEAD;
        s1 = wmma_bf16(qa0, load_frag(K1, D_HEAD), s1);
        s1 = wmma_bf16(qa1, load_frag(K1 + 32, D_HEAD), s1);

        // ---- issue Vg B-fragment loads early; latency hides under softmax ----
        v16bf vb0 = load_frag(Vh + (size_t)( 0) * SEQ + kbase, SEQ);
        v16bf vb1 = load_frag(Vh + (size_t)(16) * SEQ + kbase, SEQ);
        v16bf vb2 = load_frag(Vh + (size_t)(32) * SEQ + kbase, SEQ);
        v16bf vb3 = load_frag(Vh + (size_t)(48) * SEQ + kbase, SEQ);

        // ---- online softmax update ----
#pragma unroll
        for (int i = 0; i < 8; ++i) {
            float a = s0[i] * scale;
            float c = s1[i] * scale;
            int qrow = qrowbase + i;
            if (kbase + n15      > qrow) a = -__builtin_inff();
            if (kbase + 16 + n15 > qrow) c = -__builtin_inff();
            float mx   = rowmax16(fmaxf(a, c));
            float mnew = fmaxf(mrun[i], mx);
            float alpha = __expf(mrun[i] - mnew);
            float pa = __expf(a - mnew);
            float pc = __expf(c - mnew);
            lrun[i] = lrun[i] * alpha + rowsum16(pa + pc);
            mrun[i] = mnew;
            o[0][i] *= alpha; o[1][i] *= alpha; o[2][i] *= alpha; o[3][i] *= alpha;
            int row = i + 8 * hi;
            pw[row * 32 + n15]      = (bf16_t)pa;
            pw[row * 32 + 16 + n15] = (bf16_t)pc;
        }
        asm volatile("s_wait_dscnt 0x0" ::: "memory");  // P stores visible to lane reads

        v16bf pf = load_frag(pw, 32);
        o[0] = wmma_bf16(pf, vb0, o[0]);
        o[1] = wmma_bf16(pf, vb1, o[1]);
        o[2] = wmma_bf16(pf, vb2, o[2]);
        o[3] = wmma_bf16(pf, vb3, o[3]);
#if __has_builtin(__builtin_amdgcn_sched_group_barrier)
        __builtin_amdgcn_sched_group_barrier(0x100, 2, 0);   // pf DS reads
        __builtin_amdgcn_sched_group_barrier(0x008, 4, 0);   // 4 WMMAs
#endif
        // DS ops from one wave are in-order: next iteration's stores cannot
        // bypass this iteration's loads, so no trailing wait needed.
    }

#pragma unroll
    for (int t = 0; t < 4; ++t) {
        int col = h * D_HEAD + t * 16 + n15;
#pragma unroll
        for (int i = 0; i < 8; ++i) {
            int token = b * SEQ + qt * 16 + i + 8 * hi;
            Oa[(size_t)token * D_MODEL + col] = (bf16_t)(o[t][i] / lrun[i]);
        }
    }
}

// ---------------------------------------------------------------------------
// Output projection: (4096x1024) x (1024x1024) + bias -> fp32 d_out
// ---------------------------------------------------------------------------
__global__ void __launch_bounds__(256)
out_gemm_kernel(const bf16_t* __restrict__ Oa, const bf16_t* __restrict__ WoT,
                const float* __restrict__ bias, float* __restrict__ out) {
    __shared__ __align__(16) bf16_t Bt0[64 * KSTEP];
    __shared__ __align__(16) bf16_t Bt1[64 * KSTEP];
    int wv   = threadIdx.x >> 5;
    int lane = threadIdx.x & 31;
    int hi   = lane >> 4;
    int row0 = blockIdx.y * 128 + wv * 16;
    int col0 = blockIdx.x * 64;

    v8f acc[4] = {{}, {}, {}, {}};
    gemm_core(Oa, WoT, D_MODEL, row0, col0, Bt0, Bt1, acc);

#pragma unroll
    for (int t = 0; t < 4; ++t) {
        int col = col0 + t * 16 + (lane & 15);
        float bv = bias[col];
#pragma unroll
        for (int i = 0; i < 8; ++i) {
            int token = row0 + i + 8 * hi;
            out[(size_t)token * D_MODEL + col] = acc[t][i] + bv;
        }
    }
}

// ---------------------------------------------------------------------------
// Launch
// ---------------------------------------------------------------------------
extern "C" void kernel_launch(void* const* d_in, const int* in_sizes, int n_in,
                              void* d_out, int out_size, void* d_ws, size_t ws_size,
                              hipStream_t stream) {
    const float* x     = (const float*)d_in[0];
    const float* W_qkv = (const float*)d_in[1];
    const float* b_qkv = (const float*)d_in[2];
    const float* W_g   = (const float*)d_in[3];
    const float* W_out = (const float*)d_in[4];
    const float* b_out = (const float*)d_in[5];
    float* out = (float*)d_out;

    char* ws = (char*)d_ws;
    size_t off = 0;
    auto carve = [&](size_t bytes) { char* p = ws + off; off += (bytes + 255) & ~size_t(255); return p; };
    bf16_t* Xb    = (bf16_t*)carve((size_t)TOKENS * D_MODEL * 2);
    bf16_t* WqkvT = (bf16_t*)carve((size_t)QKV_N * D_MODEL * 2);
    bf16_t* WgT   = (bf16_t*)carve((size_t)N_HEADS * D_HEAD * D_HEAD * 2);
    bf16_t* WoT   = (bf16_t*)carve((size_t)D_MODEL * D_MODEL * 2);
    bf16_t* Qb    = (bf16_t*)carve((size_t)TOKENS * D_MODEL * 2);
    bf16_t* Kb    = (bf16_t*)carve((size_t)TOKENS * D_MODEL * 2);
    bf16_t* Vb    = (bf16_t*)carve((size_t)TOKENS * D_MODEL * 2);
    bf16_t* VgT   = (bf16_t*)carve((size_t)TOKENS * D_MODEL * 2);
    bf16_t* Oa    = (bf16_t*)carve((size_t)TOKENS * D_MODEL * 2);

    { int n = TOKENS * D_MODEL;
      cvt_f32_bf16<<<(n + 255) / 256, 256, 0, stream>>>(x, Xb, n); }
    { int n = D_MODEL * QKV_N;
      transpose_bf16<<<(n + 255) / 256, 256, 0, stream>>>(W_qkv, WqkvT, D_MODEL, QKV_N); }
    { int n = N_HEADS * D_HEAD * D_HEAD;
      wg_transpose<<<(n + 255) / 256, 256, 0, stream>>>(W_g, WgT); }
    { int n = D_MODEL * D_MODEL;
      transpose_bf16<<<(n + 255) / 256, 256, 0, stream>>>(W_out, WoT, D_MODEL, D_MODEL); }

    qkv_gemm_kernel<<<dim3(QKV_N / 64, TOKENS / 128), 256, 0, stream>>>(
        Xb, WqkvT, b_qkv, Qb, Kb, Vb);

    gate_kernel<<<dim3(BATCH * N_HEADS * (SEQ / 16) / 8), 256, 0, stream>>>(
        Qb, Vb, WgT, VgT);

    BilinearlyModulatedAttention_flash_kernel<<<dim3(BATCH * N_HEADS * (SEQ / 16) / 8),
                                                256, 0, stream>>>(Qb, Kb, VgT, Oa);

    out_gemm_kernel<<<dim3(D_MODEL / 64, TOKENS / 128), 256, 0, stream>>>(
        Oa, WoT, b_out, out);
}
